// minGRU_18193481466121
// MI455X (gfx1250) — matrix operand order repaired and checked by reading, not compiled, not run on hardware
//
#include <hip/hip_runtime.h>

#define BB 8
#define SS 8192
#define DIN 512
#define DD 512
#define CHUNK 128             // timesteps per workgroup chunk
#define NCHUNK (SS / CHUNK)   // 64 chunks along S
#define KB 256                // K block staged in LDS (2 blocks cover K=512)
#define NSTAGE 16             // 8 d-tiles x 2 K blocks
#define LDSTRIDE 264          // 256 + 8 bf16 pad -> 528B row; TDM pad 128+4 DWORDs

typedef __bf16 bf16;
typedef __attribute__((ext_vector_type(16))) __bf16 v16bf;
typedef __attribute__((ext_vector_type(8)))  float  v8f;
typedef __attribute__((ext_vector_type(4)))  unsigned int v4u;
typedef __attribute__((ext_vector_type(8)))  int  v8i;
typedef __attribute__((ext_vector_type(4)))  int  v4i;

union V16U { v16bf v; uint4 q[2]; };
union PK4 { bf16 h[4]; uint2 u; };

__device__ __forceinline__ v16bf load_frag(const bf16* p0, const bf16* p1) {
  V16U u;
  u.q[0] = *(const uint4*)p0;
  u.q[1] = *(const uint4*)p1;
  return u.v;
}

__device__ __forceinline__ float sigmoidf_(float x) {
  return 1.0f / (1.0f + __expf(-x));
}

#define HAVE_TDM __has_builtin(__builtin_amdgcn_tensor_load_to_lds)

#if HAVE_TDM
// 2D tile DMA: global (row-major, bf16) -> LDS with 264-elem padded row stride.
__device__ __forceinline__ void tdm_load_2d(unsigned lds_addr, const void* gaddr,
                                            unsigned tile_d0, unsigned tile_d1,
                                            unsigned tensor_d0, unsigned tensor_d1,
                                            unsigned stride_elems) {
  const unsigned long long ga = (unsigned long long)(uintptr_t)gaddr;
  v4u g0;
  g0[0] = 1u;                                   // count=1 valid user descriptor
  g0[1] = lds_addr;                             // LDS byte address
  g0[2] = (unsigned)(ga & 0xFFFFFFFFu);         // global_addr[31:0]
  g0[3] = (unsigned)((ga >> 32) & 0x1FFFFFFu)   // global_addr[56:32]
        | (2u << 30);                           // type = 2 ("image")
  v8i g1;
  g1[0] = (int)((1u << 16)    // data_size = 1 -> 2 bytes (bf16)
              | (1u << 20)    // pad_enable
              | (6u << 22)    // pad_interval: 128 DWORDs (=256 bf16 row)
              | (3u << 25));  // pad_amount: 4 DWORDs (=8 bf16 pad -> stride 264)
  g1[1] = (int)((tensor_d0 & 0xFFFFu) << 16);                       // dim0[15:0]
  g1[2] = (int)((tensor_d0 >> 16) | ((tensor_d1 & 0xFFFFu) << 16)); // dim0 hi | dim1 lo
  g1[3] = (int)((tensor_d1 >> 16) | (tile_d0 << 16));               // dim1 hi | tile_dim0
  g1[4] = (int)(tile_d1 & 0xFFFFu);                                 // tile_dim1; tile_dim2=0
  g1[5] = (int)stride_elems;                                        // dim0_stride[31:0]
  g1[6] = 0;
  g1[7] = 0;
  const v4i g2 = {0, 0, 0, 0};
  const v4i g3 = {0, 0, 0, 0};
#if __clang_major__ >= 23
  const v8i g4 = {0, 0, 0, 0, 0, 0, 0, 0};
  __builtin_amdgcn_tensor_load_to_lds(g0, g1, g2, g3, g4, 0);
#else
  __builtin_amdgcn_tensor_load_to_lds(g0, g1, g2, g3, 0);
#endif
}
#endif

// Prep A: x (f32) -> xbf (bf16), same layout (B*S, DIN).
__global__ __launch_bounds__(256)
void mingru_prep_x(const float* __restrict__ x, bf16* __restrict__ xbf) {
  const size_t i4 = (size_t)blockIdx.x * 256 + threadIdx.x;
  if (i4 >= ((size_t)BB * SS * DIN) / 4) return;
  const float4 v = ((const float4*)x)[i4];
  PK4 p;
  p.h[0] = (bf16)v.x; p.h[1] = (bf16)v.y; p.h[2] = (bf16)v.z; p.h[3] = (bf16)v.w;
  ((uint2*)xbf)[i4] = p.u;
}

// Prep B: W (DIN x 2D, f32) -> Wt (2D x DIN, bf16) transposed.
__global__ __launch_bounds__(256)
void mingru_prep_w(const float* __restrict__ W, bf16* __restrict__ Wt) {
  const int i4 = blockIdx.x * 256 + threadIdx.x;     // 131072 quads
  if (i4 >= (2 * DD * DIN) / 4) return;
  const int n = i4 >> 7;               // output row (col of W), 0..1023
  const int k0 = (i4 & 127) * 4;       // 4 consecutive k
  PK4 p;
#pragma unroll
  for (int j = 0; j < 4; ++j)
    p.h[j] = (bf16)W[(size_t)(k0 + j) * (2 * DD) + n];
  *(uint2*)(Wt + (size_t)n * DIN + k0) = p.u;
}

// Kernel 1: double-buffered TDM-staged bf16-WMMA GEMM (x @ W -> hidden,gate)
// + pointwise gates + intra-chunk linear scan. TDM fill of stage s+1 overlaps
// WMMA compute of stage s (per-wave TENSORcnt pipelining, wait <= 2).
__global__ __launch_bounds__(256)
void mingru_gemm_scan(const bf16* __restrict__ xbf, const bf16* __restrict__ Wt,
                      float* __restrict__ out, float* __restrict__ wsP,
                      float* __restrict__ wsA, float* __restrict__ wsV) {
  __shared__ bf16 xs0[CHUNK * LDSTRIDE];  // 66 KiB each; 264 KiB total
  __shared__ bf16 xs1[CHUNK * LDSTRIDE];  // xs1 also overlaid by a_s/v_s in epilogue
  __shared__ bf16 wt0[128 * LDSTRIDE];
  __shared__ bf16 wt1[128 * LDSTRIDE];
  float* a_s = (float*)xs1;               // [128 t][64 ch] (64 KiB <= 66 KiB)
  float* v_s = a_s + CHUNK * 64;

  const int tid  = threadIdx.x;
  const int wv   = tid >> 5;              // 8 waves (wave32)
  const int lane = tid & 31;
  const int b       = blockIdx.x >> 6;    // 0..7
  const int chunk   = blockIdx.x & 63;    // 0..63
  const int rowbase = chunk * CHUNK;

  const int colc  = lane & 15;
  const int hi16  = (lane & 16) ? 1 : 0;
  const int mrow  = wv * 16 + colc;       // wave's M tile = wv
  const int aoff  = hi16 ? 16 : 0;        // A: lanes 16-31 hold K=16..31
  const int boff  = hi16 ? 16 : 0;        // B: lanes 16-31 hold K=16..31
  const int n0    = wv * 16;              // this wave's staging slice (16 rows)

  v8f acc[8];                             // acc[0..3]=hidden subtiles, acc[4..7]=gate

  // Stage s = it*2 + kb; buffer parity = s & 1. Wave-uniform descriptors.
  auto issue_stage = [&](int s, bf16* xdst, bf16* wdst) {
    const int it2 = s >> 1;
    const int kb2 = s & 1;
    const int dbase2 = it2 * 64;
    const int row0 = (n0 < 64) ? (dbase2 + n0) : (DD + dbase2 + (n0 - 64));
    const bf16* gx = xbf + ((size_t)(b * SS + rowbase + n0) * DIN) + kb2 * KB;
    const bf16* gw = Wt + ((size_t)row0 * DIN) + kb2 * KB;
#if HAVE_TDM
    __builtin_prefetch(gw, 0, 1);         // warm GL2 (global_prefetch_b8)
    tdm_load_2d((unsigned)(uintptr_t)(xdst + n0 * LDSTRIDE), gx,
                KB, 16, DIN, (unsigned)(BB * SS), DIN);
    tdm_load_2d((unsigned)(uintptr_t)(wdst + n0 * LDSTRIDE), gw,
                KB, 16, DIN, (unsigned)(2 * DD), DIN);
#else
    for (int f = lane; f < (16 * KB) / 8; f += 32) {
      const int r = f >> 5;               // 0..15 within slice
      const int k = (f & 31) * 8;         // 0..255 step 8
      *(uint4*)(xdst + (n0 + r) * LDSTRIDE + k) = *(const uint4*)(gx + (size_t)r * DIN + k);
      *(uint4*)(wdst + (n0 + r) * LDSTRIDE + k) = *(const uint4*)(gw + (size_t)r * DIN + k);
    }
#endif
  };

  issue_stage(0, xs0, wt0);               // prologue fill of buffer 0

  for (int s = 0; s < NSTAGE; ++s) {
    const int it  = s >> 1;
    const int kb  = s & 1;
    bf16* xc = (s & 1) ? xs1 : xs0;
    bf16* wc = (s & 1) ? wt1 : wt0;

    if (kb == 0) {
#pragma unroll
      for (int j = 0; j < 8; ++j)
#pragma unroll
        for (int r = 0; r < 8; ++r)
          acc[j][r] = 0.0f;
    }

    __syncthreads();                      // prior readers of the other buffer done
    if (s + 1 < NSTAGE) {
      issue_stage(s + 1, (s & 1) ? xs0 : xs1, (s & 1) ? wt0 : wt1);
#if HAVE_TDM
      __builtin_amdgcn_s_wait_tensorcnt(2);  // stage s's 2 ops complete (in-order)
#endif
    } else {
#if HAVE_TDM
      __builtin_amdgcn_s_wait_tensorcnt(0);
#endif
    }
    __syncthreads();                      // everyone's stage-s data landed in LDS

    // 8 k-steps; per step: 1 A frag, 8 B frags, 8 WMMAs
#pragma unroll
    for (int ks = 0; ks < KB / 32; ++ks) {
      const int kk = ks * 32;
      const bf16* ap = &xc[mrow * LDSTRIDE + kk + aoff];
      const v16bf afrag = load_frag(ap, ap + 8);
      v16bf bfr[4];
#pragma unroll
      for (int j = 0; j < 4; ++j) {       // hidden subtiles n' = j*16+colc
        const bf16* bp = &wc[(j * 16 + colc) * LDSTRIDE + kk + boff];
        bfr[j] = load_frag(bp, bp + 8);
      }
#pragma unroll
      for (int j = 0; j < 4; ++j)
        acc[j] = __builtin_amdgcn_wmma_f32_16x16x32_bf16(
            false, afrag, false, bfr[j], (short)0, acc[j], false, false);
#pragma unroll
      for (int j = 0; j < 4; ++j) {       // gate subtiles n' = 64 + j*16+colc
        const bf16* bp = &wc[((4 + j) * 16 + colc) * LDSTRIDE + kk + boff];
        bfr[j] = load_frag(bp, bp + 8);
      }
#pragma unroll
      for (int j = 0; j < 4; ++j)
        acc[4 + j] = __builtin_amdgcn_wmma_f32_16x16x32_bf16(
            false, afrag, false, bfr[j], (short)0, acc[4 + j], false, false);
    }

    if (kb == 1) {                        // epilogue for d-tile `it`
      const int dbase = it * 64;
      __syncthreads();                    // all waves done reading xs1/wt1
      // Pointwise: a = sigmoid(-g), v = sigmoid(g) * g_fn(h). Overlay on xs1
      // (TDM in flight during epilogue targets buffer 0 only).
      {
        const int rbase = wv * 16 + hi16 * 8;
#pragma unroll
        for (int j = 0; j < 4; ++j) {
          const int ch = j * 16 + colc;   // 0..63
#pragma unroll
          for (int r = 0; r < 8; ++r) {
            const float h = acc[j][r];
            const float g = acc[4 + j][r];
            const float eg = __expf(g);
            const float a = 1.0f / (1.0f + eg);        // sigmoid(-g)
            const float z = 1.0f - a;                  // sigmoid(g)
            const float gh = (h >= 0.0f) ? (h + 0.5f) : sigmoidf_(h);
            const int row = rbase + r;
            a_s[row * 64 + ch] = a;
            v_s[row * 64 + ch] = z * gh;
          }
        }
      }
      __syncthreads();
      // Intra-chunk scan: p_t = a_t*p_{t-1} + v_t (p_-1=0), P_t = prod a
      if (tid < 64) {
        const int d = dbase + tid;
        float p = 0.0f, P = 1.0f;
        size_t o = (((size_t)b * SS + rowbase) * DD) + d;
        for (int t = 0; t < CHUNK; ++t) {
          const float a = a_s[t * 64 + tid];
          const float v = v_s[t * 64 + tid];
          p = a * p + v;
          P = a * P;
          out[o] = p;
          wsP[o] = P;
          o += DD;
        }
        const size_t c = (((size_t)b * NCHUNK + chunk) * DD) + d;
        wsA[c] = P;
        wsV[c] = p;
      }
    }
  }
}

// Kernel 2: scan over chunk aggregates, seeded by g(prev_hidden).
__global__ void mingru_chunk_scan(const float* __restrict__ prev,
                                  const float* __restrict__ wsA,
                                  const float* __restrict__ wsV,
                                  float* __restrict__ wsH) {
  const int idx = blockIdx.x * 256 + threadIdx.x;
  if (idx >= BB * DD) return;
  const int b = idx >> 9;
  const int d = idx & (DD - 1);
  const float ph = prev[idx];
  float h = (ph >= 0.0f) ? (ph + 0.5f) : sigmoidf_(ph);
  for (int c = 0; c < NCHUNK; ++c) {
    const size_t o = (((size_t)b * NCHUNK + c) * DD) + d;
    wsH[o] = h;                 // hidden state entering chunk c
    h = wsA[o] * h + wsV[o];
  }
}

// Kernel 3: out = p + P * H_chunkstart (float4 vectorized).
__global__ void mingru_fixup(float* __restrict__ out,
                             const float* __restrict__ wsP,
                             const float* __restrict__ wsH) {
  const size_t total4 = ((size_t)BB * SS * DD) / 4;
  const size_t i4 = (size_t)blockIdx.x * 256 + threadIdx.x;
  if (i4 >= total4) return;
  const size_t idx = i4 * 4;
  const int d = (int)(idx & (DD - 1));
  const int s = (int)((idx >> 9) & (SS - 1));
  const size_t b = idx >> 22;
  const size_t hb = ((b * NCHUNK) + (size_t)(s / CHUNK)) * DD + d;
  float4 p = ((const float4*)out)[i4];
  const float4 P = ((const float4*)wsP)[i4];
  const float4 H = *(const float4*)(wsH + hb);
  p.x += P.x * H.x; p.y += P.y * H.y;
  p.z += P.z * H.z; p.w += P.w * H.w;
  ((float4*)out)[i4] = p;
}

extern "C" void kernel_launch(void* const* d_in, const int* in_sizes, int n_in,
                              void* d_out, int out_size, void* d_ws, size_t ws_size,
                              hipStream_t stream) {
  (void)in_sizes; (void)n_in; (void)out_size; (void)ws_size;
  const float* x    = (const float*)d_in[0];   // (B,S,D_IN) f32
  const float* prev = (const float*)d_in[1];   // (B,D) f32
  const float* W    = (const float*)d_in[2];   // (D_IN, 2D) f32
  float* out = (float*)d_out;                  // (B,S,D) f32

  float* wsP = (float*)d_ws;                           // B*S*D floats (128 MiB)
  float* wsA = wsP + (size_t)BB * SS * DD;             // B*NCHUNK*D
  float* wsV = wsA + (size_t)BB * NCHUNK * DD;
  float* wsH = wsV + (size_t)BB * NCHUNK * DD;
  bf16* xbf  = (bf16*)(wsH + (size_t)BB * NCHUNK * DD); // B*S*DIN bf16 (64 MiB)
  bf16* Wtb  = xbf + (size_t)BB * SS * DIN;             // 2D x DIN bf16 (1 MiB)

  mingru_prep_x<<<(int)(((size_t)BB * SS * DIN / 4 + 255) / 256), 256, 0, stream>>>(x, xbf);
  mingru_prep_w<<<(2 * DD * DIN / 4 + 255) / 256, 256, 0, stream>>>(W, Wtb);
  mingru_gemm_scan<<<BB * NCHUNK, 256, 0, stream>>>(xbf, Wtb, out, wsP, wsA, wsV);
  mingru_chunk_scan<<<(BB * DD + 255) / 256, 256, 0, stream>>>(prev, wsA, wsV, wsH);
  mingru_fixup<<<(int)((((size_t)BB * SS * DD / 4) + 255) / 256), 256, 0, stream>>>(out, wsP, wsH);
}
